// ClippedGRU_68702296866958
// MI455X (gfx1250) — matrix operand order repaired and checked by reading, not compiled, not run on hardware
//
#include <hip/hip_runtime.h>
#include <stdint.h>

// Problem constants (from reference)
#define B_ 256
#define T_ 500
#define I_ 128
#define H_ 512
#define G_ (3 * H_)
#define CLIPV 5.0f

typedef __bf16 bf16_t;
typedef bf16_t v16bf __attribute__((ext_vector_type(16)));
typedef float  v8f   __attribute__((ext_vector_type(8)));

union Frag { v16bf v; uint4 q[2]; };

// LDS staging: 48 w_hh rows (3 gates x 16 cols) of 512 bf16 + 48 w_ih rows of 128 bf16
#define LDS_W_ELEMS (48 * H_)     // 24576 ushorts
#define LDS_V_ELEMS (48 * I_)     // 6144 ushorts
#define LDS_TOT_Q   ((LDS_W_ELEMS + LDS_V_ELEMS) / 8)  // 3840 16B chunks

#define NBLOCKS 64

#if __has_builtin(__builtin_amdgcn_global_load_async_to_lds_b128)
#define USE_ASYNC_LDS 1
typedef int v4i_t __attribute__((ext_vector_type(4)));
typedef __attribute__((address_space(1))) v4i_t* global_v4i_ptr;
typedef __attribute__((address_space(3))) v4i_t* lds_v4i_ptr;
#endif

static __device__ __forceinline__ unsigned short f2bf(float f) {
  union { float f; unsigned u; } c; c.f = f;
  unsigned u = c.u;
  unsigned r = (u + 0x7FFFu + ((u >> 16) & 1u)) >> 16;  // round-to-nearest-even
  return (unsigned short)r;
}

static __device__ __forceinline__ float sigmoidf_(float x) {
  return 1.0f / (1.0f + __expf(-x));
}

static __device__ __forceinline__ float fast_tanh(float x) {
#if __has_builtin(__builtin_amdgcn_tanhf)
  return __builtin_amdgcn_tanhf(x);
#else
  return tanhf(x);
#endif
}

static __device__ __forceinline__ void wait_async0() {
#if defined(USE_ASYNC_LDS)
#if __has_builtin(__builtin_amdgcn_s_wait_asynccnt)
  __builtin_amdgcn_s_wait_asynccnt(0);
#else
  asm volatile("s_wait_asynccnt 0" ::: "memory");
#endif
#endif
}

// ---------------- prep kernels ----------------

__global__ void cvt_bf16_kernel(const float* __restrict__ src,
                                unsigned short* __restrict__ dst, int n) {
  int i = blockIdx.x * blockDim.x + threadIdx.x;
  if (i < n) dst[i] = f2bf(src[i]);
}

__global__ void init_state_kernel(const float* __restrict__ h0,
                                  unsigned short* __restrict__ hbuf0,
                                  unsigned* __restrict__ barCnt,
                                  unsigned* __restrict__ barGen) {
  int i = blockIdx.x * blockDim.x + threadIdx.x;
  if (i < B_ * H_) hbuf0[i] = f2bf(h0[i]);
  if (i == 0) { *barCnt = 0u; *barGen = 0u; }
}

// ---------------- device-wide barrier ----------------

static __device__ __forceinline__ void grid_barrier(unsigned* cnt, unsigned* gen,
                                                    unsigned nblk) {
  __builtin_amdgcn_fence(__ATOMIC_RELEASE, "agent");
  __syncthreads();
  if (threadIdx.x == 0) {
    unsigned g = __hip_atomic_load(gen, __ATOMIC_RELAXED, __HIP_MEMORY_SCOPE_AGENT);
    unsigned prev =
        __hip_atomic_fetch_add(cnt, 1u, __ATOMIC_ACQ_REL, __HIP_MEMORY_SCOPE_AGENT);
    if (prev == nblk - 1u) {
      __hip_atomic_store(cnt, 0u, __ATOMIC_RELAXED, __HIP_MEMORY_SCOPE_AGENT);
      __hip_atomic_fetch_add(gen, 1u, __ATOMIC_RELEASE, __HIP_MEMORY_SCOPE_AGENT);
    } else {
      while (__hip_atomic_load(gen, __ATOMIC_RELAXED, __HIP_MEMORY_SCOPE_AGENT) == g) {
        __builtin_amdgcn_s_sleep(2);
      }
    }
  }
  __syncthreads();
  __builtin_amdgcn_fence(__ATOMIC_ACQUIRE, "agent");
}

// ---------------- persistent GRU kernel ----------------
// Grid: 64 blocks x 128 threads. Block = one h-column tile (htile = blk>>1).
// Each of the 4 waves owns TWO adjacent batch tiles (M=32), so every LDS B
// fragment feeds two WMMAs (ds_load : wmma = 1 : 1). Weights staged to LDS
// once via GLOBAL_LOAD_ASYNC_TO_LDS_B128; all 500 steps read B from LDS.

__launch_bounds__(128)
__global__ void gru_persistent_kernel(const unsigned short* __restrict__ xb,   // [B][T][I] bf16
                                      const unsigned short* __restrict__ wih,  // [3H][I] bf16
                                      const unsigned short* __restrict__ whh,  // [3H][H] bf16
                                      const float* __restrict__ bih,
                                      const float* __restrict__ bhh,
                                      const float* __restrict__ h0,
                                      unsigned short* __restrict__ hbuf,       // 2 x [B][H] bf16
                                      float* __restrict__ out,                 // [B][T][H]
                                      float* __restrict__ hlast,               // [B][H]
                                      int Tsteps,
                                      unsigned* __restrict__ barCnt,
                                      unsigned* __restrict__ barGen,
                                      unsigned nblk) {
  __shared__ __align__(16) unsigned short lds[LDS_W_ELEMS + LDS_V_ELEMS];

  const int lane  = threadIdx.x & 31;
  const int wave  = threadIdx.x >> 5;                 // 0..3
  const int htile = blockIdx.x >> 1;                  // 0..31 (shared by block)
  const int btile = ((blockIdx.x & 1) * 4 + wave) * 2; // 0,2,..,14 (2 tiles per wave)
  const int b0 = btile * 16;                          // rows [b0, b0+32)
  const int n0 = htile * 16;
  const int n  = lane & 15;   // A-row m and C column n
  const int hi = lane >> 4;   // lane half select
  const int ng = n0 + n;

  // ---- stage this block's weight rows into LDS (once) ----
  for (int q = threadIdx.x; q < LDS_TOT_Q; q += 128) {
    const unsigned short* src;
    if (q < LDS_W_ELEMS / 8) {
      int row = q >> 6, within = q & 63;                    // 64 chunks per w_hh row
      int grow = (row >> 4) * H_ + n0 + (row & 15);
      src = whh + (size_t)grow * H_ + within * 8;
    } else {
      int q2 = q - LDS_W_ELEMS / 8;
      int row = q2 >> 4, within = q2 & 15;                  // 16 chunks per w_ih row
      int grow = (row >> 4) * H_ + n0 + (row & 15);
      src = wih + (size_t)grow * I_ + within * 8;
    }
#if defined(USE_ASYNC_LDS)
    __builtin_amdgcn_global_load_async_to_lds_b128(
        (global_v4i_ptr)(uintptr_t)src,            // flat addr of global == AS1 addr
        (lds_v4i_ptr)&lds[q * 8],                  // addrspacecast generic -> AS3
        0, 0);
#else
    *(uint4*)&lds[q * 8] = *(const uint4*)src;
#endif
  }
  wait_async0();
  __syncthreads();

  // per-lane gate biases (C-fragment column fixed per lane)
  const float bias_r  = bih[ng] + bhh[ng];
  const float bias_z  = bih[H_ + ng] + bhh[H_ + ng];
  const float bias_xn = bih[2 * H_ + ng];
  const float bias_hn = bhh[2 * H_ + ng];

  // wave-private previous h tiles (C layout: VGPR i -> row i + 8*hi, col n)
  v8f hprev0, hprev1;
#pragma unroll
  for (int i = 0; i < 8; ++i) {
    hprev0[i] = h0[(size_t)(b0 + i + hi * 8) * H_ + ng];
    hprev1[i] = h0[(size_t)(b0 + 16 + i + hi * 8) * H_ + ng];
  }

  // LDS B-fragment bases (bytes folded into ds offsets by the compiler)
  const unsigned short* lwr = &lds[(0 * 16 + n) * H_ + hi * 16];
  const unsigned short* lwz = &lds[(1 * 16 + n) * H_ + hi * 16];
  const unsigned short* lwn = &lds[(2 * 16 + n) * H_ + hi * 16];
  const unsigned short* lvr = &lds[LDS_W_ELEMS + (0 * 16 + n) * I_ + hi * 16];
  const unsigned short* lvz = &lds[LDS_W_ELEMS + (1 * 16 + n) * I_ + hi * 16];
  const unsigned short* lvn = &lds[LDS_W_ELEMS + (2 * 16 + n) * I_ + hi * 16];

  for (int t = 0; t < Tsteps; ++t) {
    const unsigned short* hsrc = hbuf + (size_t)(t & 1) * (B_ * H_);
    unsigned short*       hdst = hbuf + (size_t)((t + 1) & 1) * (B_ * H_);
    const unsigned short* arow0 = hsrc + (size_t)(b0 + n) * H_ + hi * 16;
    const unsigned short* arow1 = hsrc + (size_t)(b0 + 16 + n) * H_ + hi * 16;
    const unsigned short* xrow0 = xb + ((size_t)(b0 + n) * T_ + t) * I_ + hi * 16;
    const unsigned short* xrow1 = xb + ((size_t)(b0 + 16 + n) * T_ + t) * I_ + hi * 16;

    v8f accR0 = {}, accZ0 = {}, accXN0 = {}, accHN0 = {};
    v8f accR1 = {}, accZ1 = {}, accXN1 = {}, accHN1 = {};

    // -------- recurrent GEMM: h(t-1) @ w_hh^T, K = 512, B from LDS --------
#pragma unroll 4
    for (int k0 = 0; k0 < H_; k0 += 32) {
      Frag a0, a1, fr, fz, fn;
      a0.q[0] = *(const uint4*)(arow0 + k0);
      a0.q[1] = *(const uint4*)(arow0 + k0 + 8);
      a1.q[0] = *(const uint4*)(arow1 + k0);
      a1.q[1] = *(const uint4*)(arow1 + k0 + 8);
      fr.q[0] = *(const uint4*)(lwr + k0); fr.q[1] = *(const uint4*)(lwr + k0 + 8);
      fz.q[0] = *(const uint4*)(lwz + k0); fz.q[1] = *(const uint4*)(lwz + k0 + 8);
      fn.q[0] = *(const uint4*)(lwn + k0); fn.q[1] = *(const uint4*)(lwn + k0 + 8);
      accR0  = __builtin_amdgcn_wmma_f32_16x16x32_bf16(false, a0.v, false, fr.v,
                                                       (short)0, accR0,  false, false);
      accR1  = __builtin_amdgcn_wmma_f32_16x16x32_bf16(false, a1.v, false, fr.v,
                                                       (short)0, accR1,  false, false);
      accZ0  = __builtin_amdgcn_wmma_f32_16x16x32_bf16(false, a0.v, false, fz.v,
                                                       (short)0, accZ0,  false, false);
      accZ1  = __builtin_amdgcn_wmma_f32_16x16x32_bf16(false, a1.v, false, fz.v,
                                                       (short)0, accZ1,  false, false);
      accHN0 = __builtin_amdgcn_wmma_f32_16x16x32_bf16(false, a0.v, false, fn.v,
                                                       (short)0, accHN0, false, false);
      accHN1 = __builtin_amdgcn_wmma_f32_16x16x32_bf16(false, a1.v, false, fn.v,
                                                       (short)0, accHN1, false, false);
    }

    // -------- fused input GEMM: x_t @ w_ih^T, K = 128, B from LDS --------
#pragma unroll
    for (int k0 = 0; k0 < I_; k0 += 32) {
      Frag a0, a1, fr, fz, fn;
      a0.q[0] = *(const uint4*)(xrow0 + k0);
      a0.q[1] = *(const uint4*)(xrow0 + k0 + 8);
      a1.q[0] = *(const uint4*)(xrow1 + k0);
      a1.q[1] = *(const uint4*)(xrow1 + k0 + 8);
      fr.q[0] = *(const uint4*)(lvr + k0); fr.q[1] = *(const uint4*)(lvr + k0 + 8);
      fz.q[0] = *(const uint4*)(lvz + k0); fz.q[1] = *(const uint4*)(lvz + k0 + 8);
      fn.q[0] = *(const uint4*)(lvn + k0); fn.q[1] = *(const uint4*)(lvn + k0 + 8);
      accR0  = __builtin_amdgcn_wmma_f32_16x16x32_bf16(false, a0.v, false, fr.v,
                                                       (short)0, accR0,  false, false);
      accR1  = __builtin_amdgcn_wmma_f32_16x16x32_bf16(false, a1.v, false, fr.v,
                                                       (short)0, accR1,  false, false);
      accZ0  = __builtin_amdgcn_wmma_f32_16x16x32_bf16(false, a0.v, false, fz.v,
                                                       (short)0, accZ0,  false, false);
      accZ1  = __builtin_amdgcn_wmma_f32_16x16x32_bf16(false, a1.v, false, fz.v,
                                                       (short)0, accZ1,  false, false);
      accXN0 = __builtin_amdgcn_wmma_f32_16x16x32_bf16(false, a0.v, false, fn.v,
                                                       (short)0, accXN0, false, false);
      accXN1 = __builtin_amdgcn_wmma_f32_16x16x32_bf16(false, a1.v, false, fn.v,
                                                       (short)0, accXN1, false, false);
    }

    // -------- gates / clip / store (both batch tiles) --------
#pragma unroll
    for (int i = 0; i < 8; ++i) {
      {
        float r  = sigmoidf_(accR0[i] + bias_r);
        float z  = sigmoidf_(accZ0[i] + bias_z);
        float nn = fast_tanh(accXN0[i] + bias_xn + r * (accHN0[i] + bias_hn));
        float hv = (1.0f - z) * nn + z * hprev0[i];
        hv = fminf(CLIPV, fmaxf(-CLIPV, hv));
        hprev0[i] = hv;
        const int brow = b0 + i + hi * 8;
        out[((size_t)brow * T_ + t) * H_ + ng] = hv;
        hdst[(size_t)brow * H_ + ng] = f2bf(hv);
      }
      {
        float r  = sigmoidf_(accR1[i] + bias_r);
        float z  = sigmoidf_(accZ1[i] + bias_z);
        float nn = fast_tanh(accXN1[i] + bias_xn + r * (accHN1[i] + bias_hn));
        float hv = (1.0f - z) * nn + z * hprev1[i];
        hv = fminf(CLIPV, fmaxf(-CLIPV, hv));
        hprev1[i] = hv;
        const int brow = b0 + 16 + i + hi * 8;
        out[((size_t)brow * T_ + t) * H_ + ng] = hv;
        hdst[(size_t)brow * H_ + ng] = f2bf(hv);
      }
    }

    grid_barrier(barCnt, barGen, nblk);
  }

  // final hidden state from register-resident hprev
#pragma unroll
  for (int i = 0; i < 8; ++i) {
    hlast[(size_t)(b0 + i + hi * 8) * H_ + ng]      = hprev0[i];
    hlast[(size_t)(b0 + 16 + i + hi * 8) * H_ + ng] = hprev1[i];
  }
}

// ---------------- host launcher ----------------

extern "C" void kernel_launch(void* const* d_in, const int* in_sizes, int n_in,
                              void* d_out, int out_size, void* d_ws, size_t ws_size,
                              hipStream_t stream) {
  const float* x   = (const float*)d_in[0];  // (B,T,I)
  const float* h0  = (const float*)d_in[1];  // (B,H)
  const float* wih = (const float*)d_in[2];  // (3H,I)
  const float* whh = (const float*)d_in[3];  // (3H,H)
  const float* bih = (const float*)d_in[4];  // (3H)
  const float* bhh = (const float*)d_in[5];  // (3H)

  float* out   = (float*)d_out;
  float* hlast = out + (size_t)B_ * T_ * H_;

  char* ws = (char*)d_ws;
  size_t off = 0;
  auto carve = [&](size_t bytes) {
    char* p = ws + off;
    off += (bytes + 255) & ~(size_t)255;
    return p;
  };
  unsigned short* xb   = (unsigned short*)carve((size_t)B_ * T_ * I_ * 2);
  unsigned short* wihb = (unsigned short*)carve((size_t)G_ * I_ * 2);
  unsigned short* whhb = (unsigned short*)carve((size_t)G_ * H_ * 2);
  unsigned short* hbuf = (unsigned short*)carve((size_t)2 * B_ * H_ * 2);
  unsigned*       bar  = (unsigned*)carve(256);

  const int nx = B_ * T_ * I_;
  cvt_bf16_kernel<<<(nx + 255) / 256, 256, 0, stream>>>(x, xb, nx);
  cvt_bf16_kernel<<<(G_ * I_ + 255) / 256, 256, 0, stream>>>(wih, wihb, G_ * I_);
  cvt_bf16_kernel<<<(G_ * H_ + 255) / 256, 256, 0, stream>>>(whh, whhb, G_ * H_);
  init_state_kernel<<<(B_ * H_ + 255) / 256, 256, 0, stream>>>(h0, hbuf, bar, bar + 1);

  gru_persistent_kernel<<<NBLOCKS, 128, 0, stream>>>(xb, wihb, whhb, bih, bhh, h0, hbuf,
                                                     out, hlast, T_, bar, bar + 1,
                                                     (unsigned)NBLOCKS);
}
